// MultiHeadAttention_79706003079877
// MI455X (gfx1250) — matrix operand order repaired and checked by reading, not compiled, not run on hardware
//
#include <hip/hip_runtime.h>
#include <hip/hip_bf16.h>

// ---------------- problem constants ----------------
static constexpr int B  = 4;
static constexpr int S  = 2048;
static constexpr int D  = 1024;
static constexpr int H  = 16;
static constexpr int HD = 64;           // head dim
static constexpr float EPS    = 1e-6f;
static constexpr float LOG2_E = 1.442695041f;

// ---------------- WMMA types ----------------
typedef __attribute__((ext_vector_type(16))) __bf16 v16bf;
typedef __attribute__((ext_vector_type(8)))  float  v8f;

union B16Frag {                 // 16 bf16 = 32 bytes = two uint4
    uint4 q[2];
    v16bf v;
};

// fp32 -> bf16 round-to-nearest-even
__device__ __forceinline__ unsigned short f2bf(float f) {
    unsigned u = __builtin_bit_cast(unsigned, f);
    u = u + 0x7FFFu + ((u >> 16) & 1u);
    return (unsigned short)(u >> 16);
}

// ---------------- tiny helper kernels ----------------
__global__ void cast_f32_to_bf16(const float* __restrict__ in,
                                 unsigned short* __restrict__ out, int n) {
    int i = blockIdx.x * blockDim.x + threadIdx.x;
    if (i < n) out[i] = f2bf(in[i]);
}

__global__ void count_masked(const unsigned char* __restrict__ mask,
                             int* __restrict__ nm) {
    __shared__ int red[256];
    int b = blockIdx.x, t = threadIdx.x;
    int c = 0;
    for (int s = t; s < S; s += 256) c += mask[b * S + s] ? 1 : 0;
    red[t] = c;
    __syncthreads();
    for (int off = 128; off > 0; off >>= 1) {
        if (t < off) red[t] += red[t + off];
        __syncthreads();
    }
    if (t == 0) nm[b] = red[0];
}

// ---------------- NT bf16 WMMA GEMM: C = A * B^T, 32x64 tile per wave ------
// A: [M,K] bf16 row-major, Bm: [N,K] bf16 row-major, C: [M,N] fp32.
// Per k-step: 4 A-chunk loads + 8 B-chunk loads feed 8 WMMAs (A fragments
// reused across 4 N-tiles, B fragments across 2 M-tiles).
// Requires M % 32 == 0, N % 64 == 0, K % 32 == 0.
__global__ __launch_bounds__(32)
void gemm_bf16_nt(const unsigned short* __restrict__ A,
                  const unsigned short* __restrict__ Bm,
                  float* __restrict__ C, int M, int N, int K) {
    const int l    = threadIdx.x;
    const int half = l >> 4;        // lane half (K-group select)
    const int lo   = l & 15;        // row (A) / col (B)
    const int m0 = blockIdx.y * 32;
    const int n0 = blockIdx.x * 64;

    const uint4* Arow0 = reinterpret_cast<const uint4*>(A + (size_t)(m0 + lo) * K);
    const uint4* Arow1 = reinterpret_cast<const uint4*>(A + (size_t)(m0 + 16 + lo) * K);
    const uint4* Brow[4];
#pragma unroll
    for (int t = 0; t < 4; ++t)
        Brow[t] = reinterpret_cast<const uint4*>(Bm + (size_t)(n0 + t * 16 + lo) * K);

    v8f acc[2][4];
#pragma unroll
    for (int mi = 0; mi < 2; ++mi)
#pragma unroll
        for (int t = 0; t < 4; ++t) acc[mi][t] = {};

    for (int k0 = 0; k0 < K; k0 += 32) {
        // A lane layout: K = {kb..kb+7} U {kb+16..kb+23}, kb = 8*half
        const int ai = (k0 >> 3) + half;
        B16Frag a0, a1;
        a0.q[0] = Arow0[ai];
        a0.q[1] = Arow0[ai + 2];
        a1.q[0] = Arow1[ai];
        a1.q[1] = Arow1[ai + 2];
        // B lane layout: K = {kb..kb+7} U {kb+16..kb+23}, kb = 8*half
        const int bi = (k0 >> 3) + half;
#pragma unroll
        for (int t = 0; t < 4; ++t) {
            B16Frag b;
            b.q[0] = Brow[t][bi];
            b.q[1] = Brow[t][bi + 2];
            acc[0][t] = __builtin_amdgcn_wmma_f32_16x16x32_bf16(
                false, a0.v, false, b.v, (short)0, acc[0][t], false, false);
            acc[1][t] = __builtin_amdgcn_wmma_f32_16x16x32_bf16(
                false, a1.v, false, b.v, (short)0, acc[1][t], false, false);
        }
    }

    // C/D layout: vgpr r -> row (tile_m0 + r + 8*half), lane lo -> col
#pragma unroll
    for (int mi = 0; mi < 2; ++mi) {
        float* Crow = C + (size_t)(m0 + mi * 16 + 8 * half) * N + n0 + lo;
#pragma unroll
        for (int t = 0; t < 4; ++t)
#pragma unroll
            for (int r = 0; r < 8; ++r)
                Crow[(size_t)r * N + t * 16] = acc[mi][t][r];
    }
}

// ---------------- RoPE + RMSNorm (+ per-dim scale) + repack ----------------
// src: [B,S,D] fp32 (column = h*HD + d); dst: [B,H,S,HD] bf16.
// One wave per (b,s,h); lane i owns rope pair (d=i, d=i+32).
__global__ __launch_bounds__(32)
void rope_rms_pack(const float* __restrict__ src,
                   unsigned short* __restrict__ dst,
                   const float* __restrict__ ln_scale,
                   const float* __restrict__ pds,   // may be null
                   const int* __restrict__ nm, int applyPds) {
    const int i   = threadIdx.x;                   // 0..31
    const int idx = blockIdx.x;
    const int h = idx % H;
    const int s = (idx / H) % S;
    const int b = idx / (H * S);

    const size_t soff = (size_t)(b * S + s) * D + h * HD;
    float first  = src[soff + i];
    float second = src[soff + 32 + i];

    // RoPE
    const float pos       = (float)(s - nm[b]);
    const float timescale = __powf(10000.f, (float)i / 32.f);
    const float ang = pos / timescale;
    float sn, cs;
    __sincosf(ang, &sn, &cs);
    const float rf = first * cs - second * sn;
    const float rs = second * cs + first * sn;

    // RMS norm over 64 dims
    float ss = rf * rf + rs * rs;
#pragma unroll
    for (int m = 16; m > 0; m >>= 1) ss += __shfl_xor(ss, m, 32);
    const float inv = rsqrtf(ss * (1.0f / 64.0f) + EPS);

    float o1 = rf * inv * ln_scale[i];
    float o2 = rs * inv * ln_scale[i + 32];
    if (applyPds) {
        const float c = LOG2_E / 8.0f;             // LOG2_E / sqrt(HD)
        o1 *= c * logf(1.f + __expf(pds[i]));
        o2 *= c * logf(1.f + __expf(pds[i + 32]));
    }

    const size_t doff = (((size_t)b * H + h) * S + s) * HD;
    dst[doff + i]      = f2bf(o1);
    dst[doff + 32 + i] = f2bf(o2);
}

// V: [B,S,D] fp32 -> [B,H,HD,S] bf16 (transposed so PV B-fragments are contiguous)
__global__ void v_transpose(const float* __restrict__ src,
                            unsigned short* __restrict__ dst) {
    size_t i = (size_t)blockIdx.x * blockDim.x + threadIdx.x;
    if (i >= (size_t)B * H * HD * S) return;
    const int s = (int)(i % S);
    const int d = (int)((i / S) % HD);
    const int h = (int)((i / ((size_t)S * HD)) % H);
    const int b = (int)(i / ((size_t)S * HD * H));
    dst[i] = f2bf(src[(size_t)(b * S + s) * D + h * HD + d]);
}

// ---------------- flash attention (one wave per 16-row Q tile) ----------------
// Q,K: [B,H,S,HD] bf16 ; Vt: [B,H,HD,S] bf16 ; Ctx: [B,S,D] bf16
__global__ __launch_bounds__(32)
void flash_attn(const unsigned short* __restrict__ Q,
                const unsigned short* __restrict__ Km,
                const unsigned short* __restrict__ Vt,
                unsigned short* __restrict__ Ctx,
                const int* __restrict__ nm) {
    __shared__ alignas(16) unsigned short p_lds[16 * 32];   // 16x32 P tile, bf16

    const int l    = threadIdx.x;
    const int half = l >> 4;
    const int lo   = l & 15;
    const int qt = blockIdx.x;
    const int bh = blockIdx.y;
    const int b = bh >> 4;          // H == 16
    const int h = bh & 15;
    const int q0 = qt * 16;
    const int nmb = nm[b];

    const unsigned short* Qbh = Q  + (size_t)bh * S * HD;
    const unsigned short* Kbh = Km + (size_t)bh * S * HD;
    const unsigned short* Vbh = Vt + (size_t)bh * HD * S;

    // Q fragments for both K=32 steps (kept in registers all loop long)
    B16Frag qa0, qa1;
    {
        const uint4* Qrow = reinterpret_cast<const uint4*>(Qbh + (size_t)(q0 + lo) * HD);
        qa0.q[0] = Qrow[2 * half];       // d in {16h..16h+7}
        qa0.q[1] = Qrow[2 * half + 1];   // d in {16h+8..16h+15}
        qa1.q[0] = Qrow[2 * half + 4];   // +32
        qa1.q[1] = Qrow[2 * half + 5];
    }

    v8f o0 = {}, o1 = {}, o2 = {}, o3 = {};
    float mrow[8], lrow[8];
#pragma unroll
    for (int r = 0; r < 8; ++r) { mrow[r] = -1e30f; lrow[r] = 0.f; }

    const int jtmax = (q0 + 15) >> 5;          // inclusive, kv tiles of 32
    for (int jt = 0; jt <= jtmax; ++jt) {
        const int jbase = jt * 32;

        // ---- scores: S = Q (16x64) * K^T -> two 16x16 tiles ----
        v8f c0 = {}, c1 = {};
        {
            const uint4* K0 = reinterpret_cast<const uint4*>(Kbh + (size_t)(jbase + lo) * HD);
            const uint4* K1 = reinterpret_cast<const uint4*>(Kbh + (size_t)(jbase + 16 + lo) * HD);
            const int kd = 2 * half;           // uint4 index of 16*half elems
            B16Frag kf;
            kf.q[0] = K0[kd];     kf.q[1] = K0[kd + 1];
            c0 = __builtin_amdgcn_wmma_f32_16x16x32_bf16(false, qa0.v, false, kf.v, (short)0, c0, false, false);
            kf.q[0] = K1[kd];     kf.q[1] = K1[kd + 1];
            c1 = __builtin_amdgcn_wmma_f32_16x16x32_bf16(false, qa0.v, false, kf.v, (short)0, c1, false, false);
            kf.q[0] = K0[kd + 4]; kf.q[1] = K0[kd + 5];   // d += 32
            c0 = __builtin_amdgcn_wmma_f32_16x16x32_bf16(false, qa1.v, false, kf.v, (short)0, c0, false, false);
            kf.q[0] = K1[kd + 4]; kf.q[1] = K1[kd + 5];
            c1 = __builtin_amdgcn_wmma_f32_16x16x32_bf16(false, qa1.v, false, kf.v, (short)0, c1, false, false);
        }

        // ---- mask + online softmax (row m = r + 8*half lives in one lane-half) ----
        const int kv0 = jbase + lo;
        const int kv1 = jbase + 16 + lo;
#pragma unroll
        for (int r = 0; r < 8; ++r) {
            const int m = q0 + r + 8 * half;
            float s0 = ((m >= kv0) && (kv0 >= nmb)) ? c0[r] : -1e38f;
            float s1 = ((m >= kv1) && (kv1 >= nmb)) ? c1[r] : -1e38f;
            float mx = fmaxf(s0, s1);
#pragma unroll
            for (int d = 8; d > 0; d >>= 1) mx = fmaxf(mx, __shfl_xor(mx, d, 32));
            const float mn    = fmaxf(mrow[r], mx);
            const float alpha = __expf(mrow[r] - mn);
            const float p0 = __expf(s0 - mn);
            const float p1 = __expf(s1 - mn);
            float rs = p0 + p1;
#pragma unroll
            for (int d = 8; d > 0; d >>= 1) rs += __shfl_xor(rs, d, 32);
            lrow[r] = lrow[r] * alpha + rs;
            mrow[r] = mn;
            o0[r] *= alpha; o1[r] *= alpha; o2[r] *= alpha; o3[r] *= alpha;

            const int row = r + 8 * half;
            p_lds[row * 32 + lo]      = f2bf(p0);
            p_lds[row * 32 + 16 + lo] = f2bf(p1);
        }
        // same-wave LDS ops are in-order; compiler inserts s_wait_dscnt.

        // ---- O += P (16x32) * V (32x64) ----
        B16Frag pa;
        {
            const uint4* Prow = reinterpret_cast<const uint4*>(&p_lds[lo * 32]);
            pa.q[0] = Prow[2 * half];
            pa.q[1] = Prow[2 * half + 1];
        }
#pragma unroll
        for (int t = 0; t < 4; ++t) {
            const uint4* Vrow = reinterpret_cast<const uint4*>(
                Vbh + (size_t)(t * 16 + lo) * S + jbase + 16 * half);
            B16Frag vf;
            vf.q[0] = Vrow[0];
            vf.q[1] = Vrow[1];
            v8f& ot = (t == 0) ? o0 : (t == 1) ? o1 : (t == 2) ? o2 : o3;
            ot = __builtin_amdgcn_wmma_f32_16x16x32_bf16(
                false, pa.v, false, vf.v, (short)0, ot, false, false);
        }
    }

    // ---- epilogue: O /= l, write context bf16 [B,S,D] ----
#pragma unroll
    for (int r = 0; r < 8; ++r) {
        const float invl = 1.0f / lrow[r];
        const size_t rowoff = (size_t)(b * S + q0 + r + 8 * half) * D + h * HD + lo;
        Ctx[rowoff +  0] = f2bf(o0[r] * invl);
        Ctx[rowoff + 16] = f2bf(o1[r] * invl);
        Ctx[rowoff + 32] = f2bf(o2[r] * invl);
        Ctx[rowoff + 48] = f2bf(o3[r] * invl);
    }
}

// ---------------- host-side orchestration ----------------
static inline size_t alignup(size_t x) { return (x + 255) & ~(size_t)255; }

extern "C" void kernel_launch(void* const* d_in, const int* in_sizes, int n_in,
                              void* d_out, int out_size, void* d_ws, size_t ws_size,
                              hipStream_t stream) {
    (void)in_sizes; (void)n_in; (void)out_size; (void)ws_size;

    const float* x   = (const float*)d_in[0];
    const float* Wq  = (const float*)d_in[1];
    const float* Wk  = (const float*)d_in[2];
    const float* Wv  = (const float*)d_in[3];
    const float* Wo  = (const float*)d_in[4];
    const float* qls = (const float*)d_in[5];
    const float* kls = (const float*)d_in[6];
    const float* pds = (const float*)d_in[7];
    const unsigned char* pmask = (const unsigned char*)d_in[8];
    float* out = (float*)d_out;

    // workspace carve-up
    char* p = (char*)d_ws;
    size_t off = 0;
    auto take = [&](size_t bytes) { char* r = p + off; off = alignup(off + bytes); return r; };

    const size_t nX = (size_t)B * S * D;          // 8,388,608
    const size_t nW = (size_t)D * D;              // 1,048,576

    unsigned short* xb  = (unsigned short*)take(nX * 2);   // x bf16; later reused as ctx bf16
    unsigned short* wqb = (unsigned short*)take(nW * 2);
    unsigned short* wkb = (unsigned short*)take(nW * 2);
    unsigned short* wvb = (unsigned short*)take(nW * 2);
    unsigned short* wob = (unsigned short*)take(nW * 2);
    float*          gf  = (float*)take(nX * 4);            // fp32 GEMM scratch (reused q->k->v)
    unsigned short* qb  = (unsigned short*)take(nX * 2);   // [B,H,S,HD]
    unsigned short* kb  = (unsigned short*)take(nX * 2);   // [B,H,S,HD]
    unsigned short* vtb = (unsigned short*)take(nX * 2);   // [B,H,HD,S]
    int*            nmp = (int*)take(B * sizeof(int));

    // 1) bf16 casts
    {
        const int T = 256;
        cast_f32_to_bf16<<<(unsigned)((nX + T - 1) / T), T, 0, stream>>>(x,  xb,  (int)nX);
        cast_f32_to_bf16<<<(unsigned)((nW + T - 1) / T), T, 0, stream>>>(Wq, wqb, (int)nW);
        cast_f32_to_bf16<<<(unsigned)((nW + T - 1) / T), T, 0, stream>>>(Wk, wkb, (int)nW);
        cast_f32_to_bf16<<<(unsigned)((nW + T - 1) / T), T, 0, stream>>>(Wv, wvb, (int)nW);
        cast_f32_to_bf16<<<(unsigned)((nW + T - 1) / T), T, 0, stream>>>(Wo, wob, (int)nW);
    }

    // 2) per-batch masked-prefix count
    count_masked<<<B, 256, 0, stream>>>(pmask, nmp);

    const dim3 gGemm(D / 64, (B * S) / 32);   // N tiles x M tiles (32x64 per wave)
    const dim3 wave(32);

    // 3) Q projection -> RoPE/RMS/per-dim-scale -> [B,H,S,HD] bf16
    gemm_bf16_nt<<<gGemm, wave, 0, stream>>>(xb, wqb, gf, B * S, D, D);
    rope_rms_pack<<<(unsigned)(B * S * H), wave, 0, stream>>>(gf, qb, qls, pds, nmp, 1);

    // 4) K projection -> RoPE/RMS
    gemm_bf16_nt<<<gGemm, wave, 0, stream>>>(xb, wkb, gf, B * S, D, D);
    rope_rms_pack<<<(unsigned)(B * S * H), wave, 0, stream>>>(gf, kb, kls, nullptr, nmp, 0);

    // 5) V projection -> transpose to [B,H,HD,S] bf16
    gemm_bf16_nt<<<gGemm, wave, 0, stream>>>(xb, wvb, gf, B * S, D, D);
    v_transpose<<<(unsigned)((nX + 255) / 256), 256, 0, stream>>>(gf, vtb);

    // 6) flash attention -> context bf16 (reuses xb; x bf16 no longer needed)
    flash_attn<<<dim3(S / 16, B * H), wave, 0, stream>>>(qb, kb, vtb, xb, nmp);

    // 7) output projection -> fp32 out
    gemm_bf16_nt<<<gGemm, wave, 0, stream>>>(xb, wob, out, B * S, D, D);
}
